// LinearBlock_13537736917144
// MI455X (gfx1250) — matrix operand order repaired and checked by reading, not compiled, not run on hardware
//
#include <hip/hip_runtime.h>
#include <math.h>

typedef __attribute__((ext_vector_type(16))) __bf16 v16bf;
typedef __attribute__((ext_vector_type(8)))  __bf16 v8bf;
typedef __attribute__((ext_vector_type(8)))  float  v8f;

#define CIN  128
#define COUT 256
#define NB   8
#define NG   8
#define EPSV 1e-5f
#define LDSW_STRIDE 136   // 128 + 8 bf16 pad -> 272B row stride, conflict-free ds_load_b128

// ---------------------------------------------------------------------------
// Convert W (fp32, [256,128] row-major) into LDS as bf16 with padded stride.
// ---------------------------------------------------------------------------
__device__ __forceinline__ void load_W_lds(const float* __restrict__ W,
                                           __bf16* ldsW, int tid) {
  const float4* W4 = (const float4*)W;  // 32768 floats = 8192 float4
  #pragma unroll
  for (int it = 0; it < 32; ++it) {
    int j = tid + it * 256;
    float4 f = W4[j];
    int i = j << 2;
    int n = i >> 7;        // W row (output channel)
    int k = i & 127;       // W col (input channel)
    __bf16* d = ldsW + n * LDSW_STRIDE + k;
    d[0] = (__bf16)f.x; d[1] = (__bf16)f.y;
    d[2] = (__bf16)f.z; d[3] = (__bf16)f.w;
  }
}

// ---------------------------------------------------------------------------
// One wave computes a 16(row) x 128(col) slab of h = feats @ W^T + bias
// as 8 WMMA 16x16 tiles, fp32 accumulators. K = 128 = 4 steps of 32.
// ---------------------------------------------------------------------------
__device__ __forceinline__ void compute_tiles(const float* __restrict__ feats,
                                              const float* __restrict__ bias,
                                              const __bf16* ldsW,
                                              int rowBase, int colBase,
                                              int lane, int N, v8f acc[8]) {
  const int mrow = lane & 15;
  const int half = lane >> 4;
  int rowA = rowBase + mrow;
  if (rowA >= N) rowA = N - 1;                       // clamp (uniform-safe)
  const float* aptr = feats + (size_t)rowA * CIN + half * 8;

  #pragma unroll
  for (int t = 0; t < 8; ++t) acc[t] = (v8f){0,0,0,0,0,0,0,0};

  #pragma unroll
  for (int ks = 0; ks < 4; ++ks) {
    const int k = ks * 32;
    // A fragment: K = k + half*8 + {0..7}  and  k + half*8 + {16..23}
    const float4* p0 = (const float4*)(aptr + k);
    float4 f0 = p0[0], f1 = p0[1];
    const float4* p1 = (const float4*)(aptr + k + 16);
    float4 f2 = p1[0], f3 = p1[1];
    v16bf a;
    a[0]=(__bf16)f0.x; a[1]=(__bf16)f0.y; a[2]=(__bf16)f0.z; a[3]=(__bf16)f0.w;
    a[4]=(__bf16)f1.x; a[5]=(__bf16)f1.y; a[6]=(__bf16)f1.z; a[7]=(__bf16)f1.w;
    a[8]=(__bf16)f2.x; a[9]=(__bf16)f2.y; a[10]=(__bf16)f2.z; a[11]=(__bf16)f2.w;
    a[12]=(__bf16)f3.x; a[13]=(__bf16)f3.y; a[14]=(__bf16)f3.z; a[15]=(__bf16)f3.w;

    #pragma unroll
    for (int t = 0; t < 8; ++t) {
      // B fragment: column n = colBase + t*16 + mrow, K = k + half*8 + {0..7} and k + half*8 + {16..23}
      const __bf16* bp = ldsW + (colBase + t * 16 + mrow) * LDSW_STRIDE
                              + k + half * 8;
      v8bf b0 = *(const v8bf*)bp;
      v8bf b1 = *(const v8bf*)(bp + 16);
      v16bf bb = __builtin_shufflevector(b0, b1,
                   0,1,2,3,4,5,6,7,8,9,10,11,12,13,14,15);
      acc[t] = __builtin_amdgcn_wmma_f32_16x16x32_bf16(
                   false, a, false, bb, (short)0, acc[t], false, false);
    }
  }
  // bias: lane's column is fixed per tile
  #pragma unroll
  for (int t = 0; t < 8; ++t) {
    float bn = bias[colBase + t * 16 + mrow];
    #pragma unroll
    for (int r = 0; r < 8; ++r) acc[t][r] += bn;
  }
}

// ---------------------------------------------------------------------------
// Kernel 0: zero stats workspace (S[64], Q[64], cnt[8], mean[64], inv[64])
// ---------------------------------------------------------------------------
__global__ void init_stats_kernel(float* ws) {
  int t = threadIdx.x;
  if (t < 264) ws[t] = 0.0f;
}

// ---------------------------------------------------------------------------
// Kernel 1: per-batch point counts
// ---------------------------------------------------------------------------
__global__ void count_kernel(const int* __restrict__ bidx, float* cnt, int N) {
  __shared__ float c[NB];
  int t = threadIdx.x;
  if (t < NB) c[t] = 0.0f;
  __syncthreads();
  int i = blockIdx.x * blockDim.x + t;
  if (i < N) atomicAdd(&c[bidx[i]], 1.0f);
  __syncthreads();
  if (t < NB) atomicAdd(&cnt[t], c[t]);
}

// ---------------------------------------------------------------------------
// Kernel 2: WMMA GEMM pass computing sum(h), sum(h^2) per (batch, group)
// ---------------------------------------------------------------------------
__global__ void __launch_bounds__(256)
stats_kernel(const float* __restrict__ feats, const int* __restrict__ bidx,
             const float* __restrict__ W, const float* __restrict__ bias,
             float* wsS, float* wsQ, int N) {
  __shared__ __bf16 ldsW[COUT * LDSW_STRIDE];
  __shared__ float ldsS[NB * NG];
  __shared__ float ldsQ[NB * NG];

  const int tid  = threadIdx.x;
  const int lane = tid & 31;
  const int wave = tid >> 5;
  if (tid < NB * NG) { ldsS[tid] = 0.0f; ldsQ[tid] = 0.0f; }
  load_W_lds(W, ldsW, tid);
  __syncthreads();

  const int rowBase = blockIdx.x * 64 + (wave & 3) * 16;
  const int colBase = (wave >> 2) * 128;

  v8f acc[8];
  compute_tiles(feats, bias, ldsW, rowBase, colBase, lane, N, acc);

  const int mrow = lane & 15;
  const int half = lane >> 4;
  const int gBase = colBase >> 5;       // 32 cols per group
  #pragma unroll
  for (int r = 0; r < 8; ++r) {
    const int row = rowBase + r + half * 8;   // C/D layout row
    #pragma unroll
    for (int gp = 0; gp < 4; ++gp) {
      float x0 = acc[2 * gp][r], x1 = acc[2 * gp + 1][r];
      float s = x0 + x1;
      float q = x0 * x0 + x1 * x1;
      #pragma unroll
      for (int m = 1; m < 16; m <<= 1) {     // reduce over 16 columns
        s += __shfl_xor(s, m, 32);
        q += __shfl_xor(q, m, 32);
      }
      if (mrow == 0 && row < N) {            // lanes 0 and 16 flush rows
        int bin = bidx[row] * NG + gBase + gp;
        atomicAdd(&ldsS[bin], s);
        atomicAdd(&ldsQ[bin], q);
      }
    }
  }
  __syncthreads();
  if (tid < NB * NG) {
    atomicAdd(&wsS[tid], ldsS[tid]);
    atomicAdd(&wsQ[tid], ldsQ[tid]);
  }
}

// ---------------------------------------------------------------------------
// Kernel 3: mean / inv-std per (batch, group)
// ---------------------------------------------------------------------------
__global__ void finalize_kernel(const float* wsS, const float* wsQ,
                                const float* wsCnt, float* wsMean, float* wsInv) {
  int t = threadIdx.x;
  if (t < NB * NG) {
    int b = t >> 3;
    float n = wsCnt[b] * (float)(COUT / NG);
    n = fmaxf(n, 1.0f);
    float mean = wsS[t] / n;
    float var  = wsQ[t] / n - mean * mean;
    wsMean[t] = mean;
    wsInv[t]  = rsqrtf(var + EPSV);
  }
}

// ---------------------------------------------------------------------------
// Kernel 4: WMMA GEMM pass 2 — normalize + affine + FiLM + SiLU, store fp32
// ---------------------------------------------------------------------------
__global__ void __launch_bounds__(256)
final_kernel(const float* __restrict__ feats, const int* __restrict__ bidx,
             const float* __restrict__ W, const float* __restrict__ bias,
             const float* __restrict__ gamma, const float* __restrict__ beta,
             const float* __restrict__ scale, const float* __restrict__ shift,
             const float* __restrict__ wsMean, const float* __restrict__ wsInv,
             float* __restrict__ out, int N) {
  __shared__ __bf16 ldsW[COUT * LDSW_STRIDE];

  const int tid  = threadIdx.x;
  const int lane = tid & 31;
  const int wave = tid >> 5;
  load_W_lds(W, ldsW, tid);
  __syncthreads();

  const int rowBase = blockIdx.x * 64 + (wave & 3) * 16;
  const int colBase = (wave >> 2) * 128;

  v8f acc[8];
  compute_tiles(feats, bias, ldsW, rowBase, colBase, lane, N, acc);

  const int mrow = lane & 15;
  const int half = lane >> 4;
  #pragma unroll
  for (int t = 0; t < 8; ++t) {
    const int n = colBase + t * 16 + mrow;
    const int g = n >> 5;
    const float gm = gamma[n], bt = beta[n];
    #pragma unroll
    for (int r = 0; r < 8; ++r) {
      const int row = rowBase + r + half * 8;
      if (row < N) {
        const int bi = bidx[row];
        const float mean = wsMean[bi * NG + g];
        const float inv  = wsInv[bi * NG + g];
        float y = (acc[t][r] - mean) * inv;
        y = y * gm + bt;
        y = y * (scale[bi * COUT + n] + 1.0f) + shift[bi * COUT + n];
        out[(size_t)row * COUT + n] = y / (1.0f + __expf(-y));   // SiLU
      }
    }
  }
}

// ---------------------------------------------------------------------------
extern "C" void kernel_launch(void* const* d_in, const int* in_sizes, int n_in,
                              void* d_out, int out_size, void* d_ws, size_t ws_size,
                              hipStream_t stream) {
  const float* feats = (const float*)d_in[0];
  const int*   bidx  = (const int*)  d_in[1];
  const float* W     = (const float*)d_in[2];
  const float* bias  = (const float*)d_in[3];
  const float* gamma = (const float*)d_in[4];
  const float* beta  = (const float*)d_in[5];
  const float* scale = (const float*)d_in[6];
  const float* shift = (const float*)d_in[7];
  float* out = (float*)d_out;
  float* ws  = (float*)d_ws;

  const int N = in_sizes[0] / CIN;
  float* wsS    = ws;          // 64
  float* wsQ    = ws + 64;     // 64
  float* wsCnt  = ws + 128;    // 8
  float* wsMean = ws + 136;    // 64
  float* wsInv  = ws + 200;    // 64

  init_stats_kernel<<<1, 256, 0, stream>>>(ws);
  count_kernel<<<(N + 255) / 256, 256, 0, stream>>>(bidx, wsCnt, N);

  const int gblocks = (N + 63) / 64;
  stats_kernel<<<gblocks, 256, 0, stream>>>(feats, bidx, W, bias, wsS, wsQ, N);
  finalize_kernel<<<1, 64, 0, stream>>>(wsS, wsQ, wsCnt, wsMean, wsInv);
  final_kernel<<<gblocks, 256, 0, stream>>>(feats, bidx, W, bias, gamma, beta,
                                            scale, shift, wsMean, wsInv, out, N);
}
